// lstm_classifier_35931696398423
// MI455X (gfx1250) — 8-bit matrix operand order repaired and checked by reading, compiled, not run on hardware
//
#include <hip/hip_runtime.h>
#include <hip/hip_bf16.h>


#define B_    256
#define T_    1000
#define F_    64
#define H1_   256
#define G1_   1024
#define H2_   128
#define G2_   512
#define DMLP_ 128
#define NCLS_ 5
#define MT    16

#define XS1   (F_ + 16)
#define HS1   (H1_ + 16)
#define XS2   (H1_ + 16)
#define HS2   (H2_ + 16)
#define G1P   (G1_ + 4)
#define G2P   (G2_ + 4)

typedef __attribute__((ext_vector_type(8))) int   v8i;
typedef __attribute__((ext_vector_type(8))) float v8f;

__device__ __forceinline__ unsigned char f2fp8(float f) {
  unsigned s = (__builtin_bit_cast(unsigned, f) >> 24) & 0x80u;
  float a = __builtin_fminf(__builtin_fabsf(f), 448.0f);
  unsigned ub = __builtin_bit_cast(unsigned, a);
  unsigned r  = ub + 0x7FFFFu + ((ub >> 20) & 1u);
  unsigned nrm = (((r >> 23) - 120u) << 3) | ((r >> 20) & 7u);
  unsigned den = (unsigned)(int)__builtin_rintf(a * 512.0f);
  unsigned mag = (a >= 0.015625f) ? nrm : den;
  return (unsigned char)(s | mag);
}

__device__ __forceinline__ float sigmoid_f(float x) {
  return __builtin_amdgcn_rcpf(1.0f + __expf(-x));
}
__device__ __forceinline__ float tanh_f(float x) {
  return 1.0f - 2.0f * __builtin_amdgcn_rcpf(__expf(2.0f * x) + 1.0f);
}

__device__ __forceinline__ v8f wmma_fp8(v8i a, v8i b, v8f c) {
  return __builtin_amdgcn_wmma_f32_16x16x64_fp8_fp8(a, b, (short)0, c, false, false);
}

__device__ __forceinline__ v8i load_A8_lds(const unsigned char* tile,
                                           int stride, int k0, int lane) {
  int m = lane & 15, h = lane >> 4;
  const unsigned char* p = tile + m * stride + k0 + 8 * h;
  v8i a;
#pragma unroll
  for (int r = 0; r < 4; ++r) {
    uint2 d = *(const uint2*)(p + 16 * r);
    a[2 * r]     = (int)d.x;
    a[2 * r + 1] = (int)d.y;
  }
  return a;
}

__device__ __forceinline__ v8i load_B8_g(const float* __restrict__ W, int Kt,
                                         int n0, int k0, int lane) {
  int nn = n0 + (lane & 15);
  int kb = k0 + 8 * (lane >> 4);
  union { v8i v; unsigned char b[32]; } u;
#pragma unroll
  for (int s = 0; s < 32; ++s) {
    u.b[s] = f2fp8(W[(long)nn * Kt + kb + 16 * (s >> 3) + (s & 7)]);
  }
  return u.v;
}

__global__ void __launch_bounds__(1024) lstm1_kernel(
    const float* __restrict__ x, const float* __restrict__ Wih1,
    const float* __restrict__ Whh1, const float* __restrict__ bih1,
    const float* __restrict__ bhh1, unsigned char* __restrict__ h1seq) {
  __shared__ float         s_gates[MT][G1P];
  __shared__ float         s_c[MT][H1_];
  __shared__ unsigned char s_h[MT][HS1];
  __shared__ unsigned char s_x[MT][XS1];
  __shared__ float         s_bias[G1_];

  const int tid = threadIdx.x;
  const int lane = tid & 31, wv = tid >> 5;
  const int m0 = blockIdx.x * MT;

  s_bias[tid] = bih1[tid] + bhh1[tid];
  for (int i = tid; i < MT * H1_; i += 1024) (&s_c[0][0])[i] = 0.0f;
  for (int i = tid; i < MT * HS1; i += 1024) (&s_h[0][0])[i] = 0;

  const int n0 = wv * 32;
  v8i wih[2], whh[2][4];
#pragma unroll
  for (int nt = 0; nt < 2; ++nt)
    wih[nt] = load_B8_g(Wih1, F_, n0 + 16 * nt, 0, lane);
#pragma unroll
  for (int nt = 0; nt < 2; ++nt)
#pragma unroll
    for (int kt = 0; kt < 4; ++kt)
      whh[nt][kt] = load_B8_g(Whh1, H1_, n0 + 16 * nt, 64 * kt, lane);
  __syncthreads();

  for (int t = 0; t < T_; ++t) {
    {
      int b = tid >> 6, f = tid & 63;
      const float* xp = &x[((long)(m0 + b) * T_ + t) * F_ + f];
      s_x[b][f] = f2fp8(*xp);
      if (t + 1 < T_) __builtin_prefetch(xp + F_, 0, 1);
    }
    __syncthreads();

    v8f acc0 = {}, acc1 = {};
    {
      v8i a = load_A8_lds(&s_x[0][0], XS1, 0, lane);
      acc0 = wmma_fp8(a, wih[0], acc0);
      acc1 = wmma_fp8(a, wih[1], acc1);
    }
#pragma unroll
    for (int kt = 0; kt < 4; ++kt) {
      v8i a = load_A8_lds(&s_h[0][0], HS1, 64 * kt, lane);
      acc0 = wmma_fp8(a, whh[0][kt], acc0);
      acc1 = wmma_fp8(a, whh[1][kt], acc1);
    }
    {
      int h = lane >> 4, nn = lane & 15;
#pragma unroll
      for (int v = 0; v < 8; ++v) {
        int m = v + 8 * h;
        s_gates[m][n0 + nn]      = acc0[v];
        s_gates[m][n0 + 16 + nn] = acc1[v];
      }
    }
    __syncthreads();

#pragma unroll
    for (int r = 0; r < 4; ++r) {
      int idx = tid + (r << 10);
      int b = idx >> 8, n = idx & 255;
      float gi = sigmoid_f(s_gates[b][n] + s_bias[n]);
      float gf = sigmoid_f(s_gates[b][n + 256] + s_bias[n + 256]);
      float gg = tanh_f(s_gates[b][n + 512] + s_bias[n + 512]);
      float go = sigmoid_f(s_gates[b][n + 768] + s_bias[n + 768]);
      float c = gf * s_c[b][n] + gi * gg;
      s_c[b][n] = c;
      unsigned char hb = f2fp8(go * tanh_f(c));
      s_h[b][n] = hb;
      h1seq[((long)(m0 + b) * T_ + t) * H1_ + n] = hb;
    }
    __syncthreads();
  }
}

__global__ void __launch_bounds__(1024) lstm2_kernel(
    const unsigned char* __restrict__ h1seq, const float* __restrict__ Wih2,
    const float* __restrict__ Whh2, const float* __restrict__ bih2,
    const float* __restrict__ bhh2, float* __restrict__ hT2) {
  __shared__ float         s_gates[MT][G2P];
  __shared__ float         s_c[MT][H2_];
  __shared__ unsigned char s_h[MT][HS2];
  __shared__ unsigned char s_x[MT][XS2];
  __shared__ float         s_bias[G2_];

  const int tid = threadIdx.x;
  const int lane = tid & 31, wv = tid >> 5;
  const int m0 = blockIdx.x * MT;

  if (tid < G2_) s_bias[tid] = bih2[tid] + bhh2[tid];
  for (int i = tid; i < MT * H2_; i += 1024) (&s_c[0][0])[i] = 0.0f;
  for (int i = tid; i < MT * HS2; i += 1024) (&s_h[0][0])[i] = 0;

  const int n0 = wv * 16;
  v8i wih[4], whh[2];
#pragma unroll
  for (int kt = 0; kt < 4; ++kt) wih[kt] = load_B8_g(Wih2, H1_, n0, 64 * kt, lane);
#pragma unroll
  for (int kt = 0; kt < 2; ++kt) whh[kt] = load_B8_g(Whh2, H2_, n0, 64 * kt, lane);
  __syncthreads();

  for (int t = 0; t < T_; ++t) {
    {
      int idx = tid << 2;
      int b = idx >> 8, n = idx & 255;
      const unsigned char* hp = &h1seq[((long)(m0 + b) * T_ + t) * H1_ + n];
      *(uchar4*)&s_x[b][n] = *(const uchar4*)hp;
      if (t + 1 < T_) __builtin_prefetch(hp + H1_, 0, 1);
    }
    __syncthreads();

    v8f acc = {};
#pragma unroll
    for (int kt = 0; kt < 4; ++kt) {
      v8i a = load_A8_lds(&s_x[0][0], XS2, 64 * kt, lane);
      acc = wmma_fp8(a, wih[kt], acc);
    }
#pragma unroll
    for (int kt = 0; kt < 2; ++kt) {
      v8i a = load_A8_lds(&s_h[0][0], HS2, 64 * kt, lane);
      acc = wmma_fp8(a, whh[kt], acc);
    }
    {
      int h = lane >> 4, nn = lane & 15;
#pragma unroll
      for (int v = 0; v < 8; ++v) s_gates[v + 8 * h][n0 + nn] = acc[v];
    }
    __syncthreads();

#pragma unroll
    for (int r = 0; r < 2; ++r) {
      int idx = tid + (r << 10);
      int b = idx >> 7, n = idx & 127;
      float gi = sigmoid_f(s_gates[b][n] + s_bias[n]);
      float gf = sigmoid_f(s_gates[b][n + 128] + s_bias[n + 128]);
      float gg = tanh_f(s_gates[b][n + 256] + s_bias[n + 256]);
      float go = sigmoid_f(s_gates[b][n + 384] + s_bias[n + 384]);
      float c = gf * s_c[b][n] + gi * gg;
      s_c[b][n] = c;
      float hh = go * tanh_f(c);
      s_h[b][n] = f2fp8(hh);
      if (t == T_ - 1) hT2[(m0 + b) * H2_ + n] = hh;
    }
    __syncthreads();
  }
}

__global__ void __launch_bounds__(128) mlp_kernel(
    const float* __restrict__ hT2, const float* __restrict__ W1,
    const float* __restrict__ b1, const float* __restrict__ W2,
    const float* __restrict__ b2, float* __restrict__ out) {
  __shared__ float sh[H2_], sz[DMLP_];
  int b = blockIdx.x, j = threadIdx.x;
  sh[j] = hT2[b * H2_ + j];
  __syncthreads();
  float acc = b1[j];
#pragma unroll 8
  for (int k = 0; k < H2_; ++k) acc = fmaf(sh[k], W1[j * H2_ + k], acc);
  sz[j] = fmaxf(acc, 0.0f);
  __syncthreads();
  if (j < NCLS_) {
    float o = b2[j];
#pragma unroll 8
    for (int k = 0; k < DMLP_; ++k) o = fmaf(sz[k], W2[j * DMLP_ + k], o);
    out[b * NCLS_ + j] = o;
  }
}

extern "C" void kernel_launch(void* const* d_in, const int* in_sizes, int n_in,
                              void* d_out, int out_size, void* d_ws,
                              size_t ws_size, hipStream_t stream) {
  const float* x    = (const float*)d_in[0];
  const float* Wih1 = (const float*)d_in[1];
  const float* Whh1 = (const float*)d_in[2];
  const float* bih1 = (const float*)d_in[3];
  const float* bhh1 = (const float*)d_in[4];
  const float* Wih2 = (const float*)d_in[5];
  const float* Whh2 = (const float*)d_in[6];
  const float* bih2 = (const float*)d_in[7];
  const float* bhh2 = (const float*)d_in[8];
  const float* W1   = (const float*)d_in[9];
  const float* b1   = (const float*)d_in[10];
  const float* W2   = (const float*)d_in[11];
  const float* b2   = (const float*)d_in[12];

  unsigned char* ws_h1 = (unsigned char*)d_ws;
  float* ws_hT2 = (float*)((char*)d_ws + (size_t)B_ * T_ * H1_ * sizeof(unsigned char));

  lstm1_kernel<<<B_ / MT, 1024, 0, stream>>>(x, Wih1, Whh1, bih1, bhh1, ws_h1);
  lstm2_kernel<<<B_ / MT, 1024, 0, stream>>>(ws_h1, Wih2, Whh2, bih2, bhh2, ws_hT2);
  mlp_kernel<<<B_, DMLP_, 0, stream>>>(ws_hT2, W1, b1, W2, b2, (float*)d_out);
}
